// Multihead_Self_Attention_18957985645078
// MI455X (gfx1250) — matrix operand order repaired and checked by reading, not compiled, not run on hardware
//
#include <hip/hip_runtime.h>
#include <hip/hip_bf16.h>

// ---------------- Problem constants (match reference) ----------------
#define BATCH 2
#define SEQ 2048
#define DM 1024
#define NH 16
#define DK 64
#define BS (BATCH * SEQ)          // 4096 rows for the big GEMMs
#define SCALE 0.125f              // 1/sqrt(64)

typedef unsigned int u32;
typedef unsigned short u16;
typedef __attribute__((ext_vector_type(16))) __bf16 v16bf;
typedef __attribute__((ext_vector_type(8)))  float  v8f;

// ---------------- bf16 helpers (bit-level, no __bf16 arithmetic) ------
__device__ __forceinline__ u16 f2bf(float f) {
  union { float f; u32 u; } c; c.f = f;
  return (u16)((c.u + 0x7FFFu + ((c.u >> 16) & 1u)) >> 16);  // RNE
}
__device__ __forceinline__ float bf2f(u16 s) {
  union { u32 u; float f; } c; c.u = ((u32)s) << 16; return c.f;
}

// ---------------- WMMA fragment loaders (contiguous k order: k = 16*(lane/16) + s) ----------
// A-matrix 16x32 bf16, row-major in memory with leading dim `ld` (elements).
// lane<16: row=lane,    K = {0..15}
// lane>=16: row=lane-16, K = {16..31}
__device__ __forceinline__ v16bf load_frag_a(const u16* base, int ld) {
  int l = threadIdx.x & 31;
  int row = l & 15;
  int kb  = (l >> 4) << 4;          // 0 or 16
  const u16* p = base + row * ld + kb;
  union { v16bf v; u32 u[8]; } f;
  const u32* q0 = (const u32*)p;          // k = kb .. kb+15
#pragma unroll
  for (int i = 0; i < 8; ++i) { f.u[i] = q0[i]; }
  return f.v;
}

// B-matrix 32x16 bf16 where B[k][n] = base[n*ld + k] (i.e. each output
// column n is a contiguous row of `base`).  lane holds column n=lane&15,
// K-half = (lane>>4)*16, 16 consecutive K elements per lane.
__device__ __forceinline__ v16bf load_frag_b(const u16* base, int ld) {
  int l = threadIdx.x & 31;
  const u16* p = base + (l & 15) * ld + ((l >> 4) << 4);
  union { v16bf v; u32 u[8]; } f;
  const u32* q = (const u32*)p;
#pragma unroll
  for (int i = 0; i < 8; ++i) f.u[i] = q[i];
  return f.v;
}

__device__ __forceinline__ v8f wmma_bf16(v16bf a, v16bf b, v8f c) {
  return __builtin_amdgcn_wmma_f32_16x16x32_bf16(false, a, false, b,
                                                 (short)0, c, false, false);
}

// half-wave (16-lane) row reductions; xor masks < 16 stay inside each half,
// matching the C/D layout where a row spans lanes {0..15} or {16..31}.
__device__ __forceinline__ float rowmax16(float v) {
  v = fmaxf(v, __shfl_xor(v, 1));
  v = fmaxf(v, __shfl_xor(v, 2));
  v = fmaxf(v, __shfl_xor(v, 4));
  v = fmaxf(v, __shfl_xor(v, 8));
  return v;
}
__device__ __forceinline__ float rowsum16(float v) {
  v += __shfl_xor(v, 1);
  v += __shfl_xor(v, 2);
  v += __shfl_xor(v, 4);
  v += __shfl_xor(v, 8);
  return v;
}

// ---------------- Kernel 0: f32 -> bf16 conversion --------------------
__global__ __launch_bounds__(256) void cvt_kernel(const float* __restrict__ src,
                                                  u16* __restrict__ dst, int n) {
  int i = blockIdx.x * 256 + threadIdx.x;
  if (i < n) dst[i] = f2bf(src[i]);
}

// ---------------- Kernel 1: fused QKV projection ----------------------
// Y = X * W^T : A = xb [BS x DM] row-major, B[k][n] = W[n*DM + k].
// Register-blocked 32x64 tile per wave: 2 A-frags x 4 B-frags -> 8 WMMAs
// per 6KB of operand traffic (21.3 FLOP/byte, L2-resident working set).
// z==0 -> Q in [B,H,S,DK]; z==1 -> K in [B,H,S,DK]; z==2 -> V^T in [B,H,DK,S]
__global__ __launch_bounds__(256) void qkv_gemm_kernel(
    const u16* __restrict__ xb,
    const u16* __restrict__ wq, const u16* __restrict__ wk,
    const u16* __restrict__ wv,
    u16* __restrict__ Q, u16* __restrict__ K, u16* __restrict__ V) {
  int wave = threadIdx.x >> 5;
  int tile = blockIdx.x * 8 + wave;       // 2048 tiles = 128 (M/32) x 16 (N/64)
  int tn = tile & 15, tm = tile >> 4;
  int z = blockIdx.z;
  const u16* W = (z == 0) ? wq : ((z == 1) ? wk : wv);
  u16* Out = (z == 0) ? Q : ((z == 1) ? K : V);

  const u16* Arow = xb + (size_t)(tm * 32) * DM;
  const u16* Brow = W + (size_t)(tn * 64) * DM;
  v8f acc[2][4] = {};
  for (int k = 0; k < DM; k += 32) {
    v16bf a0 = load_frag_a(Arow + k, DM);
    v16bf a1 = load_frag_a(Arow + (size_t)16 * DM + k, DM);
#pragma unroll
    for (int j = 0; j < 4; ++j) {
      v16bf b = load_frag_b(Brow + (size_t)(j * 16) * DM + k, DM);
      acc[0][j] = wmma_bf16(a0, b, acc[0][j]);
      acc[1][j] = wmma_bf16(a1, b, acc[1][j]);
    }
  }
  int l = threadIdx.x & 31;
  int lane15 = l & 15, halfm = (l >> 4) << 3;
#pragma unroll
  for (int i = 0; i < 2; ++i) {
#pragma unroll
    for (int j = 0; j < 4; ++j) {
      int ncol = tn * 64 + j * 16 + lane15;
      int h = ncol >> 6, d = ncol & 63;
#pragma unroll
      for (int r = 0; r < 8; ++r) {
        int m = tm * 32 + i * 16 + r + halfm;
        int b_ = m >> 11, s = m & (SEQ - 1);
        size_t idx;
        if (z == 2) idx = ((size_t)((b_ * NH + h) * DK + d)) * SEQ + s;   // V^T
        else        idx = ((size_t)((b_ * NH + h) * SEQ + s)) * DK + d;   // Q,K
        Out[idx] = f2bf(acc[i][j][r]);
      }
    }
  }
}

// ---------------- Kernel 2: RoPE on Q and K (in place) ----------------
// grid.y: 0 -> Q, 1 -> K.  One thread per (b,h,s, dim-pair i).
__global__ __launch_bounds__(256) void rope_kernel(u16* __restrict__ Q,
                                                   u16* __restrict__ K,
                                                   const int* __restrict__ pos) {
  u32 idx = blockIdx.x * 256 + threadIdx.x;   // < B*H*S*32 = 2^21
  int i  = idx & 31;                          // dim pair
  int s  = (idx >> 5) & (SEQ - 1);
  int bh = idx >> 16;
  u16* buf = (blockIdx.y == 0) ? Q : K;
  size_t base = ((size_t)bh * SEQ + s) * DK + 2 * i;
  float p = (float)pos[s];
  // theta^(-2i/DK) = exp(-(2i/DK) * ln(10000))
  float ang = p * __expf(-(2.0f * (float)i / (float)DK) * 9.210340372f);
  float c = __cosf(ang), sn = __sinf(ang);
  float e = bf2f(buf[base]), o = bf2f(buf[base + 1]);
  buf[base]     = f2bf(e * c - o * sn);
  buf[base + 1] = f2bf(e * sn + o * c);
}

// ---------------- Kernel 3: causal flash attention --------------------
// One wave handles a 16-row q-tile; 4 waves/block; per-wave private LDS
// stages P (16x32 bf16) between the softmax (C/D layout) and the PV WMMA
// (A layout).  Same-wave DS ops are in-order; s_wait_dscnt enforces it.
__global__ __launch_bounds__(128) void attn_kernel(
    const u16* __restrict__ Q, const u16* __restrict__ Kt,
    const u16* __restrict__ Vt, u16* __restrict__ AttnOut) {
  __shared__ u16 ldsP[4][16 * 32];
  int wv = threadIdx.x >> 5;
  int l  = threadIdx.x & 31;
  int qt = blockIdx.x * 4 + wv;             // 0..127 q-tile
  int bh = blockIdx.y;                      // 0..31  (b*NH + h)

  const u16* Qb = Q  + ((size_t)bh * SEQ + qt * 16) * DK;
  const u16* Kb = Kt + (size_t)bh * SEQ * DK;     // [S, DK]
  const u16* Vb = Vt + (size_t)bh * DK * SEQ;     // [DK, S]

  v16bf a0 = load_frag_a(Qb + 0, DK);
  v16bf a1 = load_frag_a(Qb + 32, DK);

  v8f o0 = {}, o1 = {}, o2 = {}, o3 = {};
  float mrow[8], lrow[8];
#pragma unroll
  for (int r = 0; r < 8; ++r) { mrow[r] = -3.0e37f; lrow[r] = 0.0f; }

  int half = l >> 4, ncol = l & 15;
  int nblocks = (qt * 16 + 15) / 32 + 1;    // causal: keys <= max row
  for (int jb = 0; jb < nblocks; ++jb) {
    int j0 = jb * 32;
    // ---- scores S = Q K^T for 32 keys (two 16-wide tiles)
    v8f s0 = {}, s1 = {};
    {
      v16bf b0 = load_frag_b(Kb + (size_t)j0 * DK + 0, DK);
      v16bf b1 = load_frag_b(Kb + (size_t)(j0 + 16) * DK + 0, DK);
      s0 = wmma_bf16(a0, b0, s0);
      s1 = wmma_bf16(a0, b1, s1);
      b0 = load_frag_b(Kb + (size_t)j0 * DK + 32, DK);
      b1 = load_frag_b(Kb + (size_t)(j0 + 16) * DK + 32, DK);
      s0 = wmma_bf16(a1, b0, s0);
      s1 = wmma_bf16(a1, b1, s1);
    }
    // ---- scale, causal mask, online softmax, stage P to LDS
#pragma unroll
    for (int r = 0; r < 8; ++r) {
      int M = r + (half << 3);
      int qrow = qt * 16 + M;
      float v0 = s0[r] * SCALE, v1 = s1[r] * SCALE;
      if (j0 + ncol > qrow)      v0 = -3.0e37f;
      if (j0 + 16 + ncol > qrow) v1 = -3.0e37f;
      float mx = rowmax16(fmaxf(v0, v1));
      float mnew = fmaxf(mrow[r], mx);
      float corr = __expf(mrow[r] - mnew);
      float e0 = __expf(v0 - mnew), e1 = __expf(v1 - mnew);
      lrow[r] = lrow[r] * corr + rowsum16(e0 + e1);
      mrow[r] = mnew;
      o0[r] *= corr; o1[r] *= corr; o2[r] *= corr; o3[r] *= corr;
      ldsP[wv][M * 32 + ncol]      = f2bf(e0);
      ldsP[wv][M * 32 + ncol + 16] = f2bf(e1);
    }
    asm volatile("s_wait_dscnt 0" ::: "memory");
    // ---- O += P * V   (P: 16x32 from LDS, V^T rows contiguous per lane)
    v16bf ap = load_frag_a((const u16*)&ldsP[wv][0], 32);
    o0 = wmma_bf16(ap, load_frag_b(Vb + (size_t)0  * SEQ + j0, SEQ), o0);
    o1 = wmma_bf16(ap, load_frag_b(Vb + (size_t)16 * SEQ + j0, SEQ), o1);
    o2 = wmma_bf16(ap, load_frag_b(Vb + (size_t)32 * SEQ + j0, SEQ), o2);
    o3 = wmma_bf16(ap, load_frag_b(Vb + (size_t)48 * SEQ + j0, SEQ), o3);
    asm volatile("s_wait_dscnt 0" ::: "memory");   // loads drain before next store
  }
  // ---- epilogue: normalize and write attn in [B,S,H*DK] for out-proj
  int b_ = bh >> 4, h = bh & 15;
#pragma unroll
  for (int r = 0; r < 8; ++r) {
    int M = r + (half << 3);
    int srow = qt * 16 + M;
    float inv = 1.0f / lrow[r];
    size_t rowoff = ((size_t)(b_ * SEQ + srow)) * DM + h * DK + ncol;
    AttnOut[rowoff + 0]  = f2bf(o0[r] * inv);
    AttnOut[rowoff + 16] = f2bf(o1[r] * inv);
    AttnOut[rowoff + 32] = f2bf(o2[r] * inv);
    AttnOut[rowoff + 48] = f2bf(o3[r] * inv);
  }
}

// ---------------- Kernel 4: output projection (fp32 out) --------------
// Same 32x64 register blocking as the QKV GEMM; writes fp32 d_out.
__global__ __launch_bounds__(256) void out_gemm_kernel(
    const u16* __restrict__ attn, const u16* __restrict__ wo,
    float* __restrict__ out) {
  int wave = threadIdx.x >> 5;
  int tile = blockIdx.x * 8 + wave;       // 2048 tiles = 128 x 16
  int tn = tile & 15, tm = tile >> 4;
  const u16* Arow = attn + (size_t)(tm * 32) * DM;
  const u16* Brow = wo + (size_t)(tn * 64) * DM;
  v8f acc[2][4] = {};
  for (int k = 0; k < DM; k += 32) {
    v16bf a0 = load_frag_a(Arow + k, DM);
    v16bf a1 = load_frag_a(Arow + (size_t)16 * DM + k, DM);
#pragma unroll
    for (int j = 0; j < 4; ++j) {
      v16bf b = load_frag_b(Brow + (size_t)(j * 16) * DM + k, DM);
      acc[0][j] = wmma_bf16(a0, b, acc[0][j]);
      acc[1][j] = wmma_bf16(a1, b, acc[1][j]);
    }
  }
  int l = threadIdx.x & 31;
  int lane15 = l & 15, halfm = (l >> 4) << 3;
#pragma unroll
  for (int i = 0; i < 2; ++i) {
#pragma unroll
    for (int j = 0; j < 4; ++j) {
      int ncol = tn * 64 + j * 16 + lane15;
#pragma unroll
      for (int r = 0; r < 8; ++r) {
        int m = tm * 32 + i * 16 + r + halfm;
        out[(size_t)m * DM + ncol] = acc[i][j][r];
      }
    }
  }
}

// ---------------- Host launcher ---------------------------------------
extern "C" void kernel_launch(void* const* d_in, const int* in_sizes, int n_in,
                              void* d_out, int out_size, void* d_ws, size_t ws_size,
                              hipStream_t stream) {
  (void)in_sizes; (void)n_in; (void)out_size; (void)ws_size;
  const float* x  = (const float*)d_in[0];
  const int*   tp = (const int*)d_in[1];
  const float* Wq = (const float*)d_in[2];
  const float* Wk = (const float*)d_in[3];
  const float* Wv = (const float*)d_in[4];
  const float* Wo = (const float*)d_in[5];
  float* out = (float*)d_out;

  // workspace carve-up (bf16 buffers), total 48 MB
  char* w = (char*)d_ws;
  u16* xb   = (u16*)w; w += (size_t)BS * DM * 2;          // 8 MB
  u16* wqb  = (u16*)w; w += (size_t)DM * DM * 2;          // 2 MB
  u16* wkb  = (u16*)w; w += (size_t)DM * DM * 2;
  u16* wvb  = (u16*)w; w += (size_t)DM * DM * 2;
  u16* wob  = (u16*)w; w += (size_t)DM * DM * 2;
  u16* Qr   = (u16*)w; w += (size_t)BS * DM * 2;          // [B,H,S,DK]
  u16* Kr   = (u16*)w; w += (size_t)BS * DM * 2;          // [B,H,S,DK]
  u16* Vt   = (u16*)w; w += (size_t)BS * DM * 2;          // [B,H,DK,S]
  u16* attn = (u16*)w; w += (size_t)BS * DM * 2;          // [B,S,DM]

  const int NX = BS * DM;       // 4194304
  const int NW = DM * DM;       // 1048576
  cvt_kernel<<<(NX + 255) / 256, 256, 0, stream>>>(x, xb, NX);
  cvt_kernel<<<(NW + 255) / 256, 256, 0, stream>>>(Wq, wqb, NW);
  cvt_kernel<<<(NW + 255) / 256, 256, 0, stream>>>(Wk, wkb, NW);
  cvt_kernel<<<(NW + 255) / 256, 256, 0, stream>>>(Wv, wvb, NW);
  cvt_kernel<<<(NW + 255) / 256, 256, 0, stream>>>(Wo, wob, NW);

  qkv_gemm_kernel<<<dim3(256, 1, 3), 256, 0, stream>>>(xb, wqb, wkb, wvb,
                                                       Qr, Kr, Vt);
  rope_kernel<<<dim3(8192, 2), 256, 0, stream>>>(Qr, Kr, tp);
  attn_kernel<<<dim3(32, 32), 128, 0, stream>>>(Qr, Kr, Vt, attn);
  out_gemm_kernel<<<256, 256, 0, stream>>>(attn, wob, out);
}
